// SteeredConvTranspose2d_78606491452291
// MI455X (gfx1250) — matrix operand order repaired and checked by reading, not compiled, not run on hardware
//
#include <hip/hip_runtime.h>
#include <math.h>

typedef __attribute__((ext_vector_type(16))) __bf16 v16bf;
typedef __attribute__((ext_vector_type(8)))  __bf16 v8bf;
typedef __attribute__((ext_vector_type(8)))  float  v8f;

#define NB      8
#define CIN     32
#define COUT    16
#define HIN     192
#define WIN     192
#define HOUT    384
#define WOUT    384
#define NMAPS   7
#define NTAPS   25

// output tile per workgroup
#define TY      32
#define TX      64
// input halo tile
#define IYT     18   // TY/2 + 2
#define IXT     34   // TX/2 + 2

#define BFRAG_ELEMS (NMAPS*NTAPS*32*16)   // 89600 bf16
#define XTILE_ELEMS (IYT*IXT*CIN)         // 19584 bf16
#define BASIS_ELEMS (15*NTAPS)            // 375 f32
#define BFRAG_BYTES (BFRAG_ELEMS*2)
#define XTILE_BYTES (XTILE_ELEMS*2)
#define SMEM_BYTES  (BFRAG_BYTES + XTILE_BYTES + BASIS_ELEMS*4)

#define ALPHA_PLANE (NB*HOUT*WOUT)

__device__ inline __bf16 f2bf(float f) {
  union { float f; unsigned u; } x; x.f = f;
  unsigned r = (x.u + 0x7fffu + ((x.u >> 16) & 1u)) >> 16;   // RNE
  union { unsigned short s; __bf16 b; } y; y.s = (unsigned short)r;
  return y.b;
}

// 15 spatial basis functions: 0..2 = rings r0={0,1,2};
// 3..8  = ring(j+1)*cos(h*th), h={1,1,2,2,3,3}; 9..14 = same with sin.
__device__ inline float basis_val(int m15, int ky, int kx) {
  float ys = (float)ky - 2.0f;
  float xs = (float)kx - 2.0f;
  float r  = sqrtf(xs*xs + ys*ys);
  float th = atan2f(ys, xs);
  float r0, ang;
  if (m15 < 3)      { r0 = (float)m15; ang = 1.0f; }
  else if (m15 < 9) { int t = m15 - 3; r0 = (float)((t & 1) + 1); ang = cosf((float)(t/2 + 1) * th); }
  else              { int t = m15 - 9; r0 = (float)((t & 1) + 1); ang = sinf((float)(t/2 + 1) * th); }
  float d = r - r0;
  return expf(-(d*d) * (1.0f / 0.72f)) * ang;     // 2*sigma^2 = 0.72
}

// One job: 2 output rows (Ya, Ya+2) of parity YPAR, 16 output pixels of
// parity XPAR each. Fully unrolled compile-time tap loop so every A/B LDS
// access is base + immediate offset.
template<int YPAR, int XPAR>
__device__ __forceinline__ void do_job(const __bf16* __restrict__ xt,
                                       const __bf16* __restrict__ Bf,
                                       const float*  __restrict__ alphag,
                                       float*        __restrict__ outg,
                                       float bv, int n, int Y0, int X0,
                                       int gg, int xsub, int lane)
{
  constexpr int NTY = YPAR ? 2 : 3;
  constexpr int NTX = XPAR ? 2 : 3;
  // kernel row/col index and LDS tile offset per tap (index [parity][tap])
  constexpr int KY[2][3]  = {{4,2,0},{3,1,1}};
  constexpr int LYO[2][3] = {{0,1,2},{1,2,2}};
  constexpr int KX[2][3]  = {{4,2,0},{3,1,1}};
  constexpr int LXO[2][3] = {{0,1,2},{1,2,2}};

  const int p    = lane & 15;          // A-matrix row (pixel) for this lane
  const int hb   = (lane & 16) >> 1;   // cin sub-block base: 0 or 8
  const int mrow = (lane >> 4) << 3;   // C-layout M offset: 0 or 8

  v8f acc0[NMAPS], acc1[NMAPS];
#pragma unroll
  for (int m = 0; m < NMAPS; ++m) {
    acc0[m] = (v8f){0,0,0,0,0,0,0,0};
    acc1[m] = (v8f){0,0,0,0,0,0,0,0};
  }

  // runtime bases; all tap/map deltas are compile-time immediates
  const __bf16* abase = &xt[(2*gg*IXT + 16*xsub + p)*CIN + hb];
  const __bf16* bbase = &Bf[lane*16];

#pragma unroll
  for (int ty = 0; ty < NTY; ++ty) {
#pragma unroll
    for (int tx = 0; tx < NTX; ++tx) {
      const __bf16* a0p = abase + (LYO[YPAR][ty]*IXT + LXO[XPAR][tx])*CIN;
      const __bf16* a1p = a0p + IXT*CIN;         // row Ya+2 -> input y + 1
      v8bf a0l = *(const v8bf*)(a0p);
      v8bf a0h = *(const v8bf*)(a0p + 16);
      v8bf a1l = *(const v8bf*)(a1p);
      v8bf a1h = *(const v8bf*)(a1p + 16);
      v16bf A0 = __builtin_shufflevector(a0l, a0h, 0,1,2,3,4,5,6,7,8,9,10,11,12,13,14,15);
      v16bf A1 = __builtin_shufflevector(a1l, a1h, 0,1,2,3,4,5,6,7,8,9,10,11,12,13,14,15);
      constexpr int dummy = 0; (void)dummy;
      const int tap = KY[YPAR][ty]*5 + KX[XPAR][tx];   // constant-folded
#pragma unroll
      for (int m = 0; m < NMAPS; ++m) {
        v16bf B = *(const v16bf*)(bbase + (m*NTAPS + tap)*512);
        acc0[m] = __builtin_amdgcn_wmma_f32_16x16x32_bf16(
                    false, A0, false, B, (short)0, acc0[m], false, false);
        acc1[m] = __builtin_amdgcn_wmma_f32_16x16x32_bf16(
                    false, A1, false, B, (short)0, acc1[m], false, false);
      }
    }
  }

  // ---- epilogue: steer + scale + bias + store ----
  const int Ya = Y0 + YPAR + 4*gg;
#pragma unroll
  for (int jr = 0; jr < 2; ++jr) {
    const v8f* acc = jr ? acc1 : acc0;
    int Y = Ya + 2*jr;
#pragma unroll
    for (int r = 0; r < 8; ++r) {
      int prow = r + mrow;                      // C layout: M = r + 8*(lane>=16)
      int Xp = X0 + 32*xsub + XPAR + 2*prow;
      int aoff = (n*HOUT + Y)*WOUT + Xp;
      float a0 = alphag[aoff];
      float a1 = alphag[ALPHA_PLANE + aoff];
      float rho = sqrtf(a0*a0 + a1*a1);
      float inv = 1.0f / (rho + 1e-8f);
      float c1 = a0*inv, s1 = a1*inv;
      float c2 = c1*c1 - s1*s1;
      float s2 = 2.0f*s1*c1;
      float c3 = c2*c1 - s2*s1;
      float s3 = s2*c1 + c2*s1;
      float v = acc[0][r] + c1*acc[1][r] + c2*acc[2][r] + c3*acc[3][r]
                          + s1*acc[4][r] + s2*acc[5][r] + s3*acc[6][r];
      v = rho*v + bv;
      outg[((n*COUT + (lane & 15))*HOUT + Y)*WOUT + Xp] = v;
    }
  }
}

__global__ __launch_bounds__(512)
void steered_convt_wmma(const float* __restrict__ xg,
                        const float* __restrict__ alphag,
                        const float* __restrict__ wg,
                        const float* __restrict__ biasg,
                        float* __restrict__ outg)
{
  extern __shared__ char smem[];
  __bf16* Bf  = (__bf16*)smem;                               // [m][tap][lane][16]
  __bf16* xt  = (__bf16*)(smem + BFRAG_BYTES);               // [ly][lx][cin]
  float*  bas = (float*)(smem + BFRAG_BYTES + XTILE_BYTES);  // [m15][tap]

  const int tid  = threadIdx.x;
  const int lane = tid & 31;
  const int wave = tid >> 5;            // 0..15
  const int n  = blockIdx.z;
  const int Y0 = blockIdx.y * TY;
  const int X0 = blockIdx.x * TX;

  // ---- stage basis table ----
  for (int e = tid; e < BASIS_ELEMS; e += blockDim.x) {
    int m15 = e / NTAPS, tap = e % NTAPS;
    bas[e] = basis_val(m15, tap / 5, tap % 5);
  }
  __syncthreads();

  // ---- stage B fragments (exact WMMA 32x16 bf16 B layout) ----
  // lane l holds column N=l&15; K = 8*(l>>4) + (s&7) + 16*(s>>3)
  for (int e = tid; e < BFRAG_ELEMS; e += blockDim.x) {
    int s   = e & 15;
    int ln  = (e >> 4) & 31;
    int tap = (e >> 9) % NTAPS;
    int m   = e / (NTAPS * 512);
    int ci  = ((ln & 16) >> 1) + (s & 7) + ((s & 8) << 1);   // K = cin
    int co  = ln & 15;                  // N = cout
    const float* wrow = wg + (co * CIN + ci) * 9;
    float acc;
    if (m == 0) {
      acc = wrow[0]*bas[0*NTAPS+tap] + wrow[1]*bas[1*NTAPS+tap] + wrow[2]*bas[2*NTAPS+tap];
    } else if (m <= 3) {                // real harmonic h=m
      int slot = 3 + 2*(m-1);
      acc = wrow[slot]*bas[slot*NTAPS+tap] + wrow[slot+1]*bas[(slot+1)*NTAPS+tap];
    } else {                            // imaginary harmonic h=m-3
      int h = m - 3;
      int slot = 3 + 2*(h-1);
      int bb   = 9 + 2*(h-1);
      acc = wrow[slot]*bas[bb*NTAPS+tap] + wrow[slot+1]*bas[(bb+1)*NTAPS+tap];
    }
    Bf[e] = f2bf(acc);
  }

  // ---- stage input halo tile as bf16, pixel-major [y][x][cin] ----
  const int y0in = Y0/2 - 1;
  const int x0in = X0/2 - 1;
  for (int e = tid; e < XTILE_ELEMS; e += blockDim.x) {
    int lx = e % IXT;
    int t2 = e / IXT;
    int ci = t2 % CIN;
    int ly = t2 / CIN;
    int iy = y0in + ly, ix = x0in + lx;
    float v = 0.0f;
    if ((unsigned)iy < (unsigned)HIN && (unsigned)ix < (unsigned)WIN)
      v = xg[((n*CIN + ci)*HIN + iy)*WIN + ix];
    xt[(ly*IXT + lx)*CIN + ci] = f2bf(v);
  }
  __syncthreads();

  const float bv = biasg[lane & 15];

  // Per-wave job slice: gg = wave&7 selects the 4-row y-group, xsub = wave>>3
  // selects the 32-wide x half. Each wave then runs all four parity classes
  // (9 + 6 + 6 + 4 = 25 tap-iterations), so every wave does equal work.
  const int gg   = wave & 7;
  const int xsub = wave >> 3;

  do_job<0,0>(xt, Bf, alphag, outg, bv, n, Y0, X0, gg, xsub, lane);
  do_job<0,1>(xt, Bf, alphag, outg, bv, n, Y0, X0, gg, xsub, lane);
  do_job<1,0>(xt, Bf, alphag, outg, bv, n, Y0, X0, gg, xsub, lane);
  do_job<1,1>(xt, Bf, alphag, outg, bv, n, Y0, X0, gg, xsub, lane);
}

extern "C" void kernel_launch(void* const* d_in, const int* in_sizes, int n_in,
                              void* d_out, int out_size, void* d_ws, size_t ws_size,
                              hipStream_t stream) {
  const float* xg     = (const float*)d_in[0];
  const float* alphag = (const float*)d_in[1];
  const float* wg     = (const float*)d_in[2];
  const float* biasg  = (const float*)d_in[3];
  float* outg = (float*)d_out;
  dim3 grid(WOUT/TX, HOUT/TY, NB);   // (6, 12, 8)
  dim3 block(512);
  steered_convt_wmma<<<grid, block, SMEM_BYTES, stream>>>(xg, alphag, wg, biasg, outg);
  (void)in_sizes; (void)n_in; (void)out_size; (void)d_ws; (void)ws_size;
}
